// GRUBase_28467043238578
// MI455X (gfx1250) — matrix operand order repaired and checked by reading, not compiled, not run on hardware
//
#include <hip/hip_runtime.h>
#include <hip/hip_bf16.h>

// GRU forward for MI455X (gfx1250).
//  Phase A (parallel prep): pack W_ih/W_hh and all of x into bf16 WMMA
//  fragment-major layouts in d_ws.
//  Phase B (recurrent): 4 persistent workgroups (16-row batch tile each),
//  16 wave32s, v_wmma_f32_16x16x32_bf16 with f32 accumulate; h carried in
//  f32 LDS across T=1024 steps; z gates carried in registers between phases.
//  Weight loads: laundered AS(1) pointers (global_load_b128, LOADcnt only,
//  no LICM hoist/spill); 1-deep software pipeline enforced with
//  sched_group_barrier so WMMAs keep a full iteration of loads in flight.

typedef __bf16 v16bf __attribute__((ext_vector_type(16)));
typedef float  v8f   __attribute__((ext_vector_type(8)));

typedef __attribute__((address_space(1))) const __bf16* gbf_ptr;
typedef __attribute__((address_space(1))) const v16bf*  gv16_ptr;

#define GRU_B 64
#define GRU_T 1024
#define GRU_I 256
#define GRU_H 512
#define GRU_CLIP 5.0f

#define XPACK_HALVES ((size_t)GRU_B * GRU_T * GRU_I)   // 16,777,216
#define WIHP_HALVES  ((size_t)1536 * 256)              //    393,216
#define WHHP_HALVES  ((size_t)1536 * 512)              //    786,432

// sched_group_barrier masks
#define SGB_WMMA      0x008
#define SGB_VMEM_READ 0x020
#define SGB_DS_READ   0x100

__device__ __forceinline__ v8f wmma_bf16(v16bf a, v16bf b, v8f c) {
  return __builtin_amdgcn_wmma_f32_16x16x32_bf16(false, a, false, b, (short)0, c,
                                                 false, false);
}

// Defeat LICM/spilling: make p look loop-variant without emitting code,
// pinned to the global address space (-> global_load_b128).
__device__ __forceinline__ gbf_ptr launder_g(const __bf16* p) {
  gbf_ptr q = (gbf_ptr)p;
  asm volatile("" : "+v"(q));
  return q;
}

// ---- A-matrix (16x32 bf16) element mapping (ISA 7.12.2) ----
__device__ __forceinline__ int afrag_idx(int m, int k) {
  int kt = k >> 5;
  int kk = k & 31;
  int lane = m + (((kk & 15) >= 8) ? 16 : 0);
  int e = ((kk & 16) ? 8 : 0) + (kk & 7);
  return ((kt << 5) + lane) * 16 + e;
}

__device__ __forceinline__ v16bf load_frag_lds(const __bf16* p, int kt, int lane) {
  return *reinterpret_cast<const v16bf*>(p + (((kt << 5) + lane) << 4));
}
__device__ __forceinline__ v16bf load_frag_g(gbf_ptr p, int fragIdx, int lane) {
  return *(gv16_ptr)(p + ((size_t)fragIdx << 9) + (lane << 4));
}

__device__ __forceinline__ float sigmoid_f(float x) {
  return 1.0f / (1.0f + __expf(-x));
}
__device__ __forceinline__ float tanh_f(float x) {
  x = fminf(fmaxf(x, -15.0f), 15.0f);
  float e = __expf(2.0f * x);
  return (e - 1.0f) / (e + 1.0f);
}
__device__ __forceinline__ float clip_f(float x) {
  return fminf(fmaxf(x, -GRU_CLIP), GRU_CLIP);
}

// ============ prep: weights -> bf16 B-fragment-major ============
__global__ void __launch_bounds__(256)
gru_pack_weights(const float* __restrict__ W_ih, const float* __restrict__ W_hh,
                 __bf16* __restrict__ wihp, __bf16* __restrict__ whhp) {
  int idx = blockIdx.x * 256 + threadIdx.x;
  if (idx < (int)WIHP_HALVES) {  // NT=96, KT=8, K=256
    int e = idx & 15, lane = (idx >> 4) & 31, kt = (idx >> 9) & 7, nt = idx >> 12;
    int g = (nt << 4) + (lane & 15);
    int k = (kt << 5) + ((e < 8) ? 0 : 16) + ((lane >> 4) << 3) + (e & 7);
    wihp[idx] = (__bf16)W_ih[(size_t)g * GRU_I + k];
  }
  if (idx < (int)WHHP_HALVES) {  // NT=96, KT=16, K=512
    int e = idx & 15, lane = (idx >> 4) & 31, kt = (idx >> 9) & 15, nt = idx >> 13;
    int g = (nt << 4) + (lane & 15);
    int k = (kt << 5) + ((e < 8) ? 0 : 16) + ((lane >> 4) << 3) + (e & 7);
    whhp[idx] = (__bf16)W_hh[(size_t)g * GRU_H + k];
  }
}

// ============ prep: x -> bf16 A-fragment-major for all t ============
__global__ void __launch_bounds__(256)
gru_pack_x(const float* __restrict__ x, __bf16* __restrict__ xpack) {
  size_t base = ((size_t)blockIdx.x * 256 + threadIdx.x) * 4;
#pragma unroll
  for (int j = 0; j < 4; ++j) {
    size_t idx = base + j;
    int e = (int)(idx & 15);
    int lane = (int)((idx >> 4) & 31);
    int kt = (int)((idx >> 9) & 7);
    int t = (int)((idx >> 12) & (GRU_T - 1));
    int bt = (int)(idx >> 22);
    int m = lane & 15;
    int k = (kt << 5) + ((e < 8) ? 0 : 16) + ((lane < 16) ? 0 : 8) + (e & 7);
    xpack[idx] = (__bf16)x[((size_t)((bt << 4) + m) * GRU_T + t) * GRU_I + k];
  }
}

// ============ recurrent persistent kernel ============
__global__ void __launch_bounds__(512)
gru_persistent_kernel(const __bf16* __restrict__ xpack, const float* __restrict__ h0,
                      const __bf16* __restrict__ wihp, const float* __restrict__ b_ih,
                      const __bf16* __restrict__ whhp, const float* __restrict__ b_hh,
                      float* __restrict__ out_hs, float* __restrict__ out_hlast) {
  extern __shared__ char smem_raw[];
  float*  hbuf   = reinterpret_cast<float*>(smem_raw);
  __bf16* hfrag  = reinterpret_cast<__bf16*>(smem_raw + 32768);
  __bf16* rhfrag = reinterpret_cast<__bf16*>(smem_raw + 49152);

  const int tid  = threadIdx.x;
  const int lane = tid & 31;
  const int w    = tid >> 5;
  const int bt   = blockIdx.x;
  const int b0   = bt << 4;

  for (int idx = tid; idx < 16 * 512; idx += 512) {
    hbuf[idx] = h0[(size_t)(b0 + (idx >> 9)) * GRU_H + (idx & 511)];
  }
  for (int idx = tid; idx < 16 * 32 * 16; idx += 512) {
    int e = idx & 15, ln = (idx >> 4) & 31, kt = idx >> 9;
    int m = ln & 15;
    int k = (kt << 5) + ((e < 8) ? 0 : 16) + ((ln < 16) ? 0 : 8) + (e & 7);
    hfrag[idx] = (__bf16)h0[(size_t)(b0 + m) * GRU_H + k];
  }
  __syncthreads();

  const int ntz = 2 * w;        // z tiles (g = 32w, 32w+16)
  const int ntr = 32 + 2 * w;   // r tiles (g = 512+32w, ...)
  const int ntn = 64 + 2 * w;   // n tiles (g = 1024+32w, ...)
  const int mbase = (lane >= 16) ? 8 : 0;

  const int c0 = (w << 5) + (lane & 15), c1 = c0 + 16;
  const float bsz0 = b_ih[c0] + b_hh[c0];
  const float bsz1 = b_ih[c1] + b_hh[c1];
  const float bsr0 = b_ih[512 + c0] + b_hh[512 + c0];
  const float bsr1 = b_ih[512 + c1] + b_hh[512 + c1];
  const float bsn0 = b_ih[1024 + c0] + b_hh[1024 + c0];
  const float bsn1 = b_ih[1024 + c1] + b_hh[1024 + c1];

  float zv0[8], zv1[8];

  for (int t = 0; t < GRU_T; ++t) {
    const int xbase = (bt * GRU_T + t) * 8;

    if (t + 1 < GRU_T) {
      __builtin_prefetch(xpack + ((size_t)(xbase + 8) << 9) + (tid << 4), 0, 1);
    }

    // ================= phase 1: z and r =================
    {
      gbf_ptr wih = launder_g(wihp);
      gbf_ptr whh = launder_g(whhp);
      gbf_ptr xp  = launder_g(xpack);

      v8f az0 = {0,0,0,0,0,0,0,0}, az1 = {0,0,0,0,0,0,0,0};
      v8f ar0 = {0,0,0,0,0,0,0,0}, ar1 = {0,0,0,0,0,0,0,0};

      // ---- x . W_ih^T, pipelined 1 deep; schedule pinned per iteration ----
      v16bf a_c  = load_frag_g(xp,  xbase,          lane);
      v16bf b0_c = load_frag_g(wih, ntz * 8,        lane);
      v16bf b1_c = load_frag_g(wih, (ntz + 1) * 8,  lane);
      v16bf b2_c = load_frag_g(wih, ntr * 8,        lane);
      v16bf b3_c = load_frag_g(wih, (ntr + 1) * 8,  lane);
#pragma unroll
      for (int kt = 0; kt < 8; ++kt) {
        v16bf a_n = a_c, b0_n = b0_c, b1_n = b1_c, b2_n = b2_c, b3_n = b3_c;
        if (kt < 7) {
          a_n  = load_frag_g(xp,  xbase + kt + 1,          lane);
          b0_n = load_frag_g(wih, ntz * 8 + kt + 1,        lane);
          b1_n = load_frag_g(wih, (ntz + 1) * 8 + kt + 1,  lane);
          b2_n = load_frag_g(wih, ntr * 8 + kt + 1,        lane);
          b3_n = load_frag_g(wih, (ntr + 1) * 8 + kt + 1,  lane);
        }
        az0 = wmma_bf16(a_c, b0_c, az0);
        az1 = wmma_bf16(a_c, b1_c, az1);
        ar0 = wmma_bf16(a_c, b2_c, ar0);
        ar1 = wmma_bf16(a_c, b3_c, ar1);
        // per-iteration schedule: next loads first, then this iter's WMMAs
        __builtin_amdgcn_sched_group_barrier(SGB_VMEM_READ, 10, 0);
        __builtin_amdgcn_sched_group_barrier(SGB_WMMA, 4, 0);
        a_c = a_n; b0_c = b0_n; b1_c = b1_n; b2_c = b2_n; b3_c = b3_n;
      }

      // ---- h . W_zr^T, pipelined 1 deep ----
      b0_c = load_frag_g(whh, ntz * 16,       lane);
      b1_c = load_frag_g(whh, (ntz + 1) * 16, lane);
      b2_c = load_frag_g(whh, ntr * 16,       lane);
      b3_c = load_frag_g(whh, (ntr + 1) * 16, lane);
#pragma unroll
      for (int kt = 0; kt < 16; ++kt) {
        v16bf b0_n = b0_c, b1_n = b1_c, b2_n = b2_c, b3_n = b3_c;
        if (kt < 15) {
          b0_n = load_frag_g(whh, ntz * 16 + kt + 1,       lane);
          b1_n = load_frag_g(whh, (ntz + 1) * 16 + kt + 1, lane);
          b2_n = load_frag_g(whh, ntr * 16 + kt + 1,       lane);
          b3_n = load_frag_g(whh, (ntr + 1) * 16 + kt + 1, lane);
        }
        v16bf a = load_frag_lds(hfrag, kt, lane);
        az0 = wmma_bf16(a, b0_c, az0);
        az1 = wmma_bf16(a, b1_c, az1);
        ar0 = wmma_bf16(a, b2_c, ar0);
        ar1 = wmma_bf16(a, b3_c, ar1);
        __builtin_amdgcn_sched_group_barrier(SGB_VMEM_READ, 8, 0);
        __builtin_amdgcn_sched_group_barrier(SGB_DS_READ, 2, 0);
        __builtin_amdgcn_sched_group_barrier(SGB_WMMA, 4, 0);
        b0_c = b0_n; b1_c = b1_n; b2_c = b2_n; b3_c = b3_n;
      }

#pragma unroll
      for (int v = 0; v < 8; ++v) {
        int m = v + mbase;
        zv0[v] = sigmoid_f(az0[v] + bsz0);
        zv1[v] = sigmoid_f(az1[v] + bsz1);
        float r0 = sigmoid_f(ar0[v] + bsr0);
        float r1 = sigmoid_f(ar1[v] + bsr1);
        rhfrag[afrag_idx(m, c0)] = (__bf16)(r0 * hbuf[m * 512 + c0]);
        rhfrag[afrag_idx(m, c1)] = (__bf16)(r1 * hbuf[m * 512 + c1]);
      }
    }
    __syncthreads();

    // ================= phase 2: n gate + state update =================
    {
      gbf_ptr wih = launder_g(wihp);
      gbf_ptr whh = launder_g(whhp);
      gbf_ptr xp  = launder_g(xpack);

      v8f an0 = {0,0,0,0,0,0,0,0}, an1 = {0,0,0,0,0,0,0,0};

      v16bf a_c  = load_frag_g(xp,  xbase,         lane);
      v16bf b0_c = load_frag_g(wih, ntn * 8,       lane);
      v16bf b1_c = load_frag_g(wih, (ntn + 1) * 8, lane);
#pragma unroll
      for (int kt = 0; kt < 8; ++kt) {
        v16bf a_n = a_c, b0_n = b0_c, b1_n = b1_c;
        if (kt < 7) {
          a_n  = load_frag_g(xp,  xbase + kt + 1,         lane);
          b0_n = load_frag_g(wih, ntn * 8 + kt + 1,       lane);
          b1_n = load_frag_g(wih, (ntn + 1) * 8 + kt + 1, lane);
        }
        an0 = wmma_bf16(a_c, b0_c, an0);
        an1 = wmma_bf16(a_c, b1_c, an1);
        __builtin_amdgcn_sched_group_barrier(SGB_VMEM_READ, 6, 0);
        __builtin_amdgcn_sched_group_barrier(SGB_WMMA, 2, 0);
        a_c = a_n; b0_c = b0_n; b1_c = b1_n;
      }

      b0_c = load_frag_g(whh, ntn * 16,       lane);
      b1_c = load_frag_g(whh, (ntn + 1) * 16, lane);
#pragma unroll
      for (int kt = 0; kt < 16; ++kt) {
        v16bf b0_n = b0_c, b1_n = b1_c;
        if (kt < 15) {
          b0_n = load_frag_g(whh, ntn * 16 + kt + 1,       lane);
          b1_n = load_frag_g(whh, (ntn + 1) * 16 + kt + 1, lane);
        }
        v16bf a = load_frag_lds(rhfrag, kt, lane);
        an0 = wmma_bf16(a, b0_c, an0);
        an1 = wmma_bf16(a, b1_c, an1);
        __builtin_amdgcn_sched_group_barrier(SGB_VMEM_READ, 4, 0);
        __builtin_amdgcn_sched_group_barrier(SGB_DS_READ, 2, 0);
        __builtin_amdgcn_sched_group_barrier(SGB_WMMA, 2, 0);
        b0_c = b0_n; b1_c = b1_n;
      }

#pragma unroll
      for (int v = 0; v < 8; ++v) {
        int m = v + mbase;
        float nv0 = tanh_f(an0[v] + bsn0);
        float nv1 = tanh_f(an1[v] + bsn1);
        float ho0 = hbuf[m * 512 + c0];
        float ho1 = hbuf[m * 512 + c1];
        float hn0 = clip_f(zv0[v] * ho0 + (1.0f - zv0[v]) * nv0);
        float hn1 = clip_f(zv1[v] * ho1 + (1.0f - zv1[v]) * nv1);
        size_t orow = ((size_t)(b0 + m) * GRU_T + t) * GRU_H;
        out_hs[orow + c0] = hn0;
        out_hs[orow + c1] = hn1;
        hbuf[m * 512 + c0] = hn0;
        hbuf[m * 512 + c1] = hn1;
        hfrag[afrag_idx(m, c0)] = (__bf16)hn0;
        hfrag[afrag_idx(m, c1)] = (__bf16)hn1;
      }
    }
    __syncthreads();
  }

  for (int idx = tid; idx < 16 * 512; idx += 512) {
    out_hlast[(size_t)(b0 + (idx >> 9)) * GRU_H + (idx & 511)] = hbuf[idx];
  }
}

extern "C" void kernel_launch(void* const* d_in, const int* in_sizes, int n_in,
                              void* d_out, int out_size, void* d_ws, size_t ws_size,
                              hipStream_t stream) {
  (void)in_sizes; (void)n_in; (void)out_size; (void)ws_size;
  const float* x    = (const float*)d_in[0];
  const float* h0   = (const float*)d_in[1];
  const float* W_ih = (const float*)d_in[2];
  const float* b_ih = (const float*)d_in[3];
  const float* W_hh = (const float*)d_in[4];
  const float* b_hh = (const float*)d_in[5];
  float* out_hs    = (float*)d_out;
  float* out_hlast = out_hs + (size_t)GRU_B * GRU_T * GRU_H;

  __bf16* xpack = (__bf16*)d_ws;                 // 32 MiB
  __bf16* wihp  = xpack + XPACK_HALVES;          // 768 KiB
  __bf16* whhp  = wihp + WIHP_HALVES;            // 1.5 MiB

  gru_pack_weights<<<dim3((unsigned)((WHHP_HALVES + 255) / 256)), dim3(256), 0,
                     stream>>>(W_ih, W_hh, wihp, whhp);
  gru_pack_x<<<dim3((unsigned)(XPACK_HALVES / (256 * 4))), dim3(256), 0, stream>>>(
      x, xpack);

  gru_persistent_kernel<<<dim3(GRU_B / 16), dim3(512), 65536, stream>>>(
      xpack, h0, wihp, b_ih, whhp, b_hh, out_hs, out_hlast);
}
